// MultiHeadedAttention_48180943127307
// MI455X (gfx1250) — matrix operand order repaired and checked by reading, not compiled, not run on hardware
//
#include <hip/hip_runtime.h>

typedef unsigned short u16;
typedef __bf16 v16bf __attribute__((ext_vector_type(16)));
typedef float  v8f   __attribute__((ext_vector_type(8)));

union Frag16 { v16bf v; u16 u[16]; uint4 q[2]; };
union Pack8  { uint4 q; u16 u[8]; };

constexpr int Bc = 8, Sc = 1024, Dc = 1024, Hc = 16, DKc = 64;

__device__ __forceinline__ u16 f32_to_bf16(float f) {
  unsigned int u = __float_as_uint(f);
  u += 0x7FFFu + ((u >> 16) & 1u);          // round-to-nearest-even
  return (u16)(u >> 16);
}
__device__ __forceinline__ float bf16_to_f32(u16 h) {
  return __uint_as_float(((unsigned int)h) << 16);
}

// ---- CDNA5 async global->LDS copy (no VGPR round trip, ASYNCcnt-tracked) ----
// LDS byte address = low 32 bits of the generic pointer (ISA 10.2 LDS aperture).
__device__ __forceinline__ void async_copy_b128(void* lds_dst, const void* gsrc) {
  unsigned lds_off = (unsigned)(unsigned long long)lds_dst;
  asm volatile("global_load_async_to_lds_b128 %0, %1, off"
               :: "v"(lds_off), "v"(gsrc) : "memory");
}
__device__ __forceinline__ void wait_async() {
  asm volatile("s_wait_asynccnt 0x0" ::: "memory");
}

// 16 contiguous bf16 -> fragment (2 x ds_load_b128)
__device__ __forceinline__ v16bf load_frag_contig(const u16* p) {
  Frag16 f;
  f.q[0] = *(const uint4*)(p);
  f.q[1] = *(const uint4*)(p + 8);
  return f.v;
}
// two 8-element contiguous chunks -> fragment (A-matrix pattern)
__device__ __forceinline__ v16bf load_frag_split(const u16* p0, const u16* p1) {
  Frag16 f;
  f.q[0] = *(const uint4*)(p0);
  f.q[1] = *(const uint4*)(p1);
  return f.v;
}

// ---------------- elementwise f32 -> bf16 -----------------------------------
__global__ void cvt_f32_bf16_kernel(const float* __restrict__ in, u16* __restrict__ out, int n) {
  int i = blockIdx.x * blockDim.x + threadIdx.x;
  if (i < n) out[i] = f32_to_bf16(in[i]);
}

// ---------------- f32 [K][N] -> bf16 transposed [N][K] (tiled) --------------
__global__ __launch_bounds__(256) void cvt_transpose_bf16_kernel(
    const float* __restrict__ in, u16* __restrict__ out, int K, int N) {
  __shared__ float t[32][33];
  int k0 = blockIdx.y * 32, n0 = blockIdx.x * 32;
  int tx = threadIdx.x & 31, ty = threadIdx.x >> 5;   // ty 0..7
#pragma unroll
  for (int i = 0; i < 4; ++i) {
    int r = ty + i * 8;
    t[r][tx] = in[(size_t)(k0 + r) * N + n0 + tx];
  }
  __syncthreads();
#pragma unroll
  for (int i = 0; i < 4; ++i) {
    int r = ty + i * 8;
    out[(size_t)(n0 + r) * K + k0 + tx] = f32_to_bf16(t[tx][r]);
  }
}

// ---------------- generic bf16 WMMA GEMM (B pre-transposed) -----------------
// C = (A[MxK] @ B[KxN] + bias) * scale, with BT = B^T stored [N][K]
// out_mode 0: f32 row-major to outF
// out_mode 1: bf16 head-split [B,H,S,64] to outH   (row=b*S+s, col=h*64+d)
// out_mode 2: bf16 transposed head-split [B,H,64,S] to outH
// Double-buffered LDS, staged with async global->LDS copies.
__global__ __launch_bounds__(128) void gemm_bf16_kernel(
    const u16* __restrict__ A, int lda,
    const u16* __restrict__ BT, int ldbt,
    const float* __restrict__ bias, float scale,
    int M, int N, int K, int out_mode,
    float* __restrict__ outF, u16* __restrict__ outH) {
  __shared__ u16 As[2][64][32];   // [buf][m][k]
  __shared__ u16 Bs[2][64][32];   // [buf][n][k]  (N-major -> B frag from two 8-wide k chunks)
  int tid = threadIdx.x, lane = tid & 31, wave = tid >> 5;
  int m0 = blockIdx.y * 64, n0 = blockIdx.x * 64;
  const int mA = lane & 15;
  const int kh = (lane >> 4) << 3;
  const int nB = wave * 16 + (lane & 15);
  const int kB = (lane >> 4) << 4;

  auto issue_tiles = [&](int k0, int buf) {
    const u16* Ag = A + (size_t)m0 * lda + k0;
    const u16* Bg = BT + (size_t)n0 * ldbt + k0;
#pragma unroll
    for (int gi = 0; gi < 2; ++gi) {
      int g = tid + gi * 128;
      int r = g >> 2, c = (g & 3) << 3;
      async_copy_b128(&As[buf][r][c], &Ag[(size_t)r * lda + c]);
      async_copy_b128(&Bs[buf][r][c], &Bg[(size_t)r * ldbt + c]);
    }
  };

  v8f acc[4] = {};
  int nk = K >> 5;
  issue_tiles(0, 0);
  for (int ki = 0; ki < nk; ++ki) {
    wait_async();
    __syncthreads();
    if (ki + 1 < nk) issue_tiles((ki + 1) << 5, (ki + 1) & 1);
    int buf = ki & 1;
    v16bf bf = load_frag_split(&Bs[buf][nB][kh], &Bs[buf][nB][kh + 16]);
#pragma unroll
    for (int mt = 0; mt < 4; ++mt) {
      int m = mt * 16 + mA;
      v16bf af = load_frag_split(&As[buf][m][kh], &As[buf][m][kh + 16]);
      acc[mt] = __builtin_amdgcn_wmma_f32_16x16x32_bf16(false, af, false, bf,
                                                        (short)0, acc[mt], false, false);
    }
  }
#pragma unroll
  for (int mt = 0; mt < 4; ++mt) {
#pragma unroll
    for (int j = 0; j < 8; ++j) {
      int row = m0 + mt * 16 + j + ((lane >> 4) << 3);
      int col = n0 + wave * 16 + (lane & 15);
      float val = acc[mt][j];
      if (bias) val += bias[col];
      val *= scale;
      if (out_mode == 0) {
        outF[(size_t)row * N + col] = val;
      } else {
        int b = row / Sc, s = row % Sc;
        int h = col >> 6, d = col & 63;
        if (out_mode == 1)
          outH[(((size_t)(b * Hc + h) * Sc + s) << 6) + d] = f32_to_bf16(val);
        else
          outH[(((size_t)(b * Hc + h) << 6) + d) * Sc + s] = f32_to_bf16(val);
      }
    }
  }
}

// ---------------- P[row, v] = dot(q_row(scaled), rel_k[v]) ------------------
__global__ void relk_dot_kernel(const u16* __restrict__ qh, const float* __restrict__ rel_k,
                                float* __restrict__ P, int total) {
  int i = blockIdx.x * blockDim.x + threadIdx.x;
  if (i >= total) return;
  int row = i / 65, v = i - row * 65;
  const uint4* q4 = (const uint4*)(qh + ((size_t)row << 6));
  const float* r = rel_k + v * 64;
  float acc = 0.f;
#pragma unroll
  for (int c = 0; c < 8; ++c) {
    Pack8 p; p.q = q4[c];
#pragma unroll
    for (int e = 0; e < 8; ++e) acc += bf16_to_f32(p.u[e]) * r[c * 8 + e];
  }
  P[i] = acc;
}

// ---------------- scores = q.k^T + rel-key gather, masked -> attn region ----
__global__ __launch_bounds__(128) void scores_kernel(
    const u16* __restrict__ qh, const u16* __restrict__ kh,
    const float* __restrict__ P, const unsigned char* __restrict__ mask,
    const float* __restrict__ adj, float* __restrict__ attn) {
  __shared__ u16 Qs[64][64];   // [q][d]
  __shared__ u16 Ks[64][64];   // [kcol][d]  (d contiguous -> B frag from two 8-wide d chunks)
  int tid = threadIdx.x, lane = tid & 31, wave = tid >> 5;
  int kt = blockIdx.x, qt = blockIdx.y, bh = blockIdx.z;
  int b = bh >> 4;
  const u16* qg = qh + ((size_t)bh * Sc + qt * 64) * 64;
  const u16* kg = kh + ((size_t)bh * Sc + kt * 64) * 64;
#pragma unroll
  for (int gi = 0; gi < 4; ++gi) {
    int g = tid + gi * 128;
    int r = g >> 3, c = (g & 7) << 3;
    async_copy_b128(&Qs[r][c], &qg[((size_t)r << 6) + c]);
    async_copy_b128(&Ks[r][c], &kg[((size_t)r << 6) + c]);
  }
  wait_async();
  __syncthreads();
  const int mA = lane & 15;
  const int kh2 = (lane >> 4) << 3;
  const int nB = wave * 16 + (lane & 15);
  const int kB = (lane >> 4) << 4;
  v8f acc[4] = {};
#pragma unroll
  for (int ks = 0; ks < 2; ++ks) {
    v16bf bf = load_frag_split(&Ks[nB][ks * 32 + kh2], &Ks[nB][ks * 32 + kh2 + 16]);
#pragma unroll
    for (int mt = 0; mt < 4; ++mt) {
      int m = mt * 16 + mA;
      v16bf af = load_frag_split(&Qs[m][ks * 32 + kh2], &Qs[m][ks * 32 + kh2 + 16]);
      acc[mt] = __builtin_amdgcn_wmma_f32_16x16x32_bf16(false, af, false, bf,
                                                        (short)0, acc[mt], false, false);
    }
  }
#pragma unroll
  for (int mt = 0; mt < 4; ++mt) {
#pragma unroll
    for (int j = 0; j < 8; ++j) {
      int qrow = qt * 64 + mt * 16 + j + ((lane >> 4) << 3);
      int kcol = kt * 64 + wave * 16 + (lane & 15);
      int d = kcol - qrow;
      d = d < -32 ? -32 : (d > 32 ? 32 : d);
      float val = acc[mt][j] + P[((size_t)bh * Sc + qrow) * 65 + (d + 32)];
      size_t idx = ((size_t)bh * Sc + qrow) * Sc + kcol;
      if (mask[((size_t)b * Sc + qrow) * Sc + kcol]) val = -1e18f;
      if (adj[idx] == 0.f) val = -1e18f;
      attn[idx] = val;
    }
  }
}

// ---------------- fused softmax + adj renorm + L1 + rel-value buckets -------
__global__ __launch_bounds__(128) void softmax_kernel(
    float* __restrict__ attn, const float* __restrict__ adj, float* __restrict__ Aws) {
  __shared__ float red[128];
  int row = blockIdx.x;              // bh*S + q
  int q   = row & (Sc - 1);
  int tid = threadIdx.x;
  size_t base = (size_t)row * Sc;
  float v[8];
  float mx = -INFINITY;
#pragma unroll
  for (int i = 0; i < 8; ++i) { v[i] = attn[base + tid + i * 128]; mx = fmaxf(mx, v[i]); }
  red[tid] = mx; __syncthreads();
  for (int o = 64; o; o >>= 1) { if (tid < o) red[tid] = fmaxf(red[tid], red[tid + o]); __syncthreads(); }
  mx = red[0]; __syncthreads();
  float sum = 0.f;
#pragma unroll
  for (int i = 0; i < 8; ++i) { v[i] = __expf(v[i] - mx); sum += v[i]; }
  red[tid] = sum; __syncthreads();
  for (int o = 64; o; o >>= 1) { if (tid < o) red[tid] += red[tid + o]; __syncthreads(); }
  float inv = 1.f / red[0]; __syncthreads();
  float l1 = 0.f;
#pragma unroll
  for (int i = 0; i < 8; ++i) {
    float a = adj[base + tid + i * 128];
    float w = v[i] * inv * (a != 0.f ? 1.f / a : 1e-18f);
    v[i] = w; l1 += w;
  }
  red[tid] = l1; __syncthreads();
  for (int o = 64; o; o >>= 1) { if (tid < o) red[tid] += red[tid + o]; __syncthreads(); }
  float invl = 1.f / fmaxf(red[0], 1e-12f); __syncthreads();
  float* Arow = Aws + (size_t)row * 65;
  if (tid < 65) Arow[tid] = 0.f;
  __syncthreads();
  float lo = 0.f, hi = 0.f;
#pragma unroll
  for (int i = 0; i < 8; ++i) {
    int k = tid + i * 128;
    float w = v[i] * invl;
    attn[base + k] = w;
    int d = k - q;
    if (d <= -32) lo += w;
    else if (d >= 32) hi += w;
    else Arow[d + 32] = w;           // unique writer per bucket
  }
  red[tid] = lo; __syncthreads();
  for (int o = 64; o; o >>= 1) { if (tid < o) red[tid] += red[tid + o]; __syncthreads(); }
  if (tid == 0) Arow[0] = red[0]; __syncthreads();
  red[tid] = hi; __syncthreads();
  for (int o = 64; o; o >>= 1) { if (tid < o) red[tid] += red[tid + o]; __syncthreads(); }
  if (tid == 0) Arow[64] = red[0];
}

// ---------------- ctx = attn @ V + A @ rel_v (merged-head layout) -----------
// vhT layout: [B,H,DV,S]
__global__ __launch_bounds__(128) void context_kernel(
    const float* __restrict__ attn, const u16* __restrict__ vhT,
    const float* __restrict__ Aws, const float* __restrict__ rel_v,
    float* __restrict__ ctx) {
  __shared__ u16 Ats[64][64];   // [q][k]
  __shared__ u16 Vs[64][64];    // [dv][k]  (k contiguous -> B frag from two 8-wide k chunks)
  int tid = threadIdx.x, lane = tid & 31, wave = tid >> 5;
  int qt = blockIdx.y, bh = blockIdx.z;
  const float* arow = attn + ((size_t)bh * Sc + qt * 64) * Sc;
  const u16* vbase = vhT + ((size_t)bh << 6) * Sc;
  const int mA = lane & 15;
  const int kh2 = (lane >> 4) << 3;
  const int nB = wave * 16 + (lane & 15);
  const int kB = (lane >> 4) << 4;
  v8f acc[4] = {};
  for (int kt = 0; kt < Sc / 64; ++kt) {
    // async V tile copy first: overlaps with attn f32->bf16 conversion below
#pragma unroll
    for (int gi = 0; gi < 4; ++gi) {
      int g = tid + gi * 128;
      int r = g >> 3, c = (g & 7) << 3;
      async_copy_b128(&Vs[r][c], &vbase[(size_t)r * Sc + kt * 64 + c]);
    }
    if (kt + 1 < Sc / 64)
      __builtin_prefetch(&arow[(size_t)(tid >> 1) * Sc + kt * 64 + 64], 0, 3);
#pragma unroll
    for (int gi = 0; gi < 4; ++gi) {
      int g = tid + gi * 128;
      int r = g >> 3, c = (g & 7) << 3;
      const float* src = &arow[(size_t)r * Sc + kt * 64 + c];
      Pack8 p;
#pragma unroll
      for (int e = 0; e < 8; ++e) p.u[e] = f32_to_bf16(src[e]);
      *(uint4*)(&Ats[r][c]) = p.q;
    }
    wait_async();
    __syncthreads();
#pragma unroll
    for (int ks = 0; ks < 2; ++ks) {
      v16bf bf = load_frag_split(&Vs[nB][ks * 32 + kh2], &Vs[nB][ks * 32 + kh2 + 16]);
#pragma unroll
      for (int mt = 0; mt < 4; ++mt) {
        int m = mt * 16 + mA;
        v16bf af = load_frag_split(&Ats[m][ks * 32 + kh2], &Ats[m][ks * 32 + kh2 + 16]);
        acc[mt] = __builtin_amdgcn_wmma_f32_16x16x32_bf16(false, af, false, bf,
                                                          (short)0, acc[mt], false, false);
      }
    }
    __syncthreads();
  }
  int bI = bh >> 4, h = bh & 15;
  int dv = wave * 16 + (lane & 15);
#pragma unroll
  for (int mt = 0; mt < 4; ++mt) {
#pragma unroll
    for (int j = 0; j < 8; ++j) {
      int qrow = qt * 64 + mt * 16 + j + ((lane >> 4) << 3);
      const float* Ar = Aws + ((size_t)bh * Sc + qrow) * 65;
      float val = acc[mt][j];
      for (int v = 0; v < 65; ++v) val += Ar[v] * rel_v[v * 64 + dv];
      ctx[(size_t)(bI * Sc + qrow) * (Hc * 64) + h * 64 + dv] = val;
    }
  }
}

// ---------------------------------------------------------------------------
extern "C" void kernel_launch(void* const* d_in, const int* in_sizes, int n_in,
                              void* d_out, int out_size, void* d_ws, size_t ws_size,
                              hipStream_t stream) {
  (void)in_sizes; (void)n_in; (void)out_size; (void)ws_size;
  const float* key   = (const float*)d_in[0];
  const float* value = (const float*)d_in[1];
  const float* query = (const float*)d_in[2];
  const unsigned char* mask = (const unsigned char*)d_in[3];
  const float* adj   = (const float*)d_in[4];
  const float* Wq = (const float*)d_in[5];
  const float* bq = (const float*)d_in[6];
  const float* Wk = (const float*)d_in[7];
  const float* bk = (const float*)d_in[8];
  const float* Wv = (const float*)d_in[9];
  const float* bv = (const float*)d_in[10];
  const float* Wo = (const float*)d_in[11];
  const float* bo = (const float*)d_in[12];
  const float* rel_k = (const float*)d_in[13];
  const float* rel_v = (const float*)d_in[14];

  float* outF = (float*)d_out;
  float* attn = outF + (size_t)Bc * Sc * Dc;   // [B,H,S,S] region of d_out

  char* w = (char*)d_ws;
  auto carve = [&](size_t bytes) -> char* {
    char* p = w; w += (bytes + 255) & ~(size_t)255; return p;
  };
  const size_t nBSD  = (size_t)Bc * Sc * Dc;          // 8M
  const size_t nW    = (size_t)Dc * (Hc * DKc);       // 1M
  const size_t nHead = (size_t)Bc * Hc * Sc * DKc;    // 8M
  u16* qryb = (u16*)carve(nBSD * 2);
  u16* keyb = (u16*)carve(nBSD * 2);
  u16* valb = (u16*)carve(nBSD * 2);
  u16* WqT  = (u16*)carve(nW * 2);
  u16* WkT  = (u16*)carve(nW * 2);
  u16* WvT  = (u16*)carve(nW * 2);
  u16* WoT  = (u16*)carve(nW * 2);
  u16* qhp  = (u16*)carve(nHead * 2);
  u16* khp  = (u16*)carve(nHead * 2);
  u16* vhT  = (u16*)carve(nHead * 2);
  float* Pp  = (float*)carve((size_t)Bc * Hc * Sc * 65 * 4);
  float* Ap  = (float*)carve((size_t)Bc * Hc * Sc * 65 * 4);
  float* ctx = (float*)carve(nBSD * 4);
  u16* ctxb  = (u16*)carve(nBSD * 2);

  // 1) conversions (weights transposed for N-major LDS B tiles)
  cvt_f32_bf16_kernel<<<(nBSD + 255) / 256, 256, 0, stream>>>(query, qryb, (int)nBSD);
  cvt_f32_bf16_kernel<<<(nBSD + 255) / 256, 256, 0, stream>>>(key,   keyb, (int)nBSD);
  cvt_f32_bf16_kernel<<<(nBSD + 255) / 256, 256, 0, stream>>>(value, valb, (int)nBSD);
  dim3 gT((Hc * DKc) / 32, Dc / 32);
  cvt_transpose_bf16_kernel<<<gT, 256, 0, stream>>>(Wq, WqT, Dc, Hc * DKc);
  cvt_transpose_bf16_kernel<<<gT, 256, 0, stream>>>(Wk, WkT, Dc, Hc * DKc);
  cvt_transpose_bf16_kernel<<<gT, 256, 0, stream>>>(Wv, WvT, Dc, Hc * DKc);
  cvt_transpose_bf16_kernel<<<dim3(Dc / 32, (Hc * DKc) / 32), 256, 0, stream>>>(Wo, WoT, Hc * DKc, Dc);

  // 2) projections (1/sqrt(DK) folded into Q; V written transposed)
  dim3 blk(128);
  dim3 gProj((Hc * DKc) / 64, (Bc * Sc) / 64);
  gemm_bf16_kernel<<<gProj, blk, 0, stream>>>(qryb, Dc, WqT, Dc, bq, 0.125f,
                                              Bc * Sc, Hc * DKc, Dc, 1, nullptr, qhp);
  gemm_bf16_kernel<<<gProj, blk, 0, stream>>>(keyb, Dc, WkT, Dc, bk, 1.0f,
                                              Bc * Sc, Hc * DKc, Dc, 1, nullptr, khp);
  gemm_bf16_kernel<<<gProj, blk, 0, stream>>>(valb, Dc, WvT, Dc, bv, 1.0f,
                                              Bc * Sc, Hc * DKc, Dc, 2, nullptr, vhT);

  // 3) rel-key dot table
  int totP = Bc * Hc * Sc * 65;
  relk_dot_kernel<<<(totP + 255) / 256, 256, 0, stream>>>(qhp, rel_k, Pp, totP);

  // 4) scores -> attn region of d_out
  dim3 gScore(Sc / 64, Sc / 64, Bc * Hc);
  scores_kernel<<<gScore, blk, 0, stream>>>(qhp, khp, Pp, mask, adj, attn);

  // 5) fused softmax / adj renorm / L1 / rel-value buckets (in-place)
  softmax_kernel<<<Bc * Hc * Sc, 128, 0, stream>>>(attn, adj, Ap);

  // 6) context
  dim3 gCtx(1, Sc / 64, Bc * Hc);
  context_kernel<<<gCtx, blk, 0, stream>>>(attn, vhT, Ap, rel_v, ctx);

  // 7) output projection
  cvt_f32_bf16_kernel<<<(nBSD + 255) / 256, 256, 0, stream>>>(ctx, ctxb, (int)nBSD);
  dim3 gOut(Dc / 64, (Bc * Sc) / 64);
  gemm_bf16_kernel<<<gOut, blk, 0, stream>>>(ctxb, Hc * DKc, WoT, Hc * DKc, bo, 1.0f,
                                             Bc * Sc, Dc, Hc * DKc, 0, outF, nullptr);
}
